// QuantizedLinear_85607288143950
// MI455X (gfx1250) — matrix operand order repaired and checked by reading, not compiled, not run on hardware
//
#include <hip/hip_runtime.h>

// QuantizedLinear: out[m,n] = scale[n] * (x[m,:] . (q[n,:] - zp[n]))
// M=8192 (4*2048), N=4096, K=4096.
// Strategy: f16 WMMA GEMM (v_wmma_f32_16x16x32_f16), q kept exact in f16,
// zero-point folded in as one extra K-tile (x_extra=1, B_extra=-zp[n]),
// scale applied in the f32 epilogue.

typedef __attribute__((ext_vector_type(16))) _Float16 v16h;
typedef __attribute__((ext_vector_type(8)))  _Float16 v8h;
typedef __attribute__((ext_vector_type(8)))  float    v8f;

#define K_DIM 4096
#define N_DIM 4096
#define BM 128
#define BN 128
#define BK 32
#define LDS_PITCH 40   // 32 data halves + 8 pad; 80B row stride = 16B-aligned, bank-conflict-free

__global__ __launch_bounds__(256, 2)
void qlinear_wmma_kernel(const float* __restrict__ x,
                         const int*   __restrict__ wq,
                         const float* __restrict__ scale,
                         const float* __restrict__ zp,
                         float*       __restrict__ out)
{
    __shared__ __align__(16) _Float16 As[BM * LDS_PITCH];  // 10 KB
    __shared__ __align__(16) _Float16 Bs[BN * LDS_PITCH];  // 10 KB

    const int t    = threadIdx.x;
    const int wave = t >> 5;
    const int lane = t & 31;
    const int lm   = lane & 15;   // fragment row/col within 16
    const int lh   = lane >> 4;   // half-wave select

    const int m_off = (wave >> 2) * 64;   // 0 / 64     : wave M offset in block tile
    const int n_off = (wave & 3)  * 32;   // 0/32/64/96 : wave N offset in block tile

    const int m_base = blockIdx.y * BM;
    const int n_base = blockIdx.x * BN;

    // Cooperative tile-load coordinates: 256 threads cover 128 rows x 2 segments of 16.
    const int lrow = t >> 1;             // 0..127
    const int lseg = (t & 1) * 16;       // 0 or 16

    const float* xptr = x  + (size_t)(m_base + lrow) * K_DIM + lseg;
    const int*   qptr = wq + (size_t)(n_base + lrow) * K_DIM + lseg;

    v8f acc[4][2] = {};

    const int NKT = K_DIM / BK;          // 128 real K-tiles + 1 zero-point tile

    for (int kt = 0; kt <= NKT; ++kt) {
        v8h a0 = {}, a1 = {}, b0 = {}, b1 = {};

        if (kt < NKT) {
            const float* xp = xptr + kt * BK;
            const int*   qp = qptr + kt * BK;

            float4 f0 = *(const float4*)(xp + 0);
            float4 f1 = *(const float4*)(xp + 4);
            float4 f2 = *(const float4*)(xp + 8);
            float4 f3 = *(const float4*)(xp + 12);
            int4   q0 = *(const int4*)(qp + 0);
            int4   q1 = *(const int4*)(qp + 4);
            int4   q2 = *(const int4*)(qp + 8);
            int4   q3 = *(const int4*)(qp + 12);

            if (kt + 1 < NKT) {                       // global_prefetch_b8 next K-tile
                __builtin_prefetch(xp + BK, 0, 3);
                __builtin_prefetch(qp + BK, 0, 3);
            }

            a0[0] = (_Float16)f0.x; a0[1] = (_Float16)f0.y;
            a0[2] = (_Float16)f0.z; a0[3] = (_Float16)f0.w;
            a0[4] = (_Float16)f1.x; a0[5] = (_Float16)f1.y;
            a0[6] = (_Float16)f1.z; a0[7] = (_Float16)f1.w;
            a1[0] = (_Float16)f2.x; a1[1] = (_Float16)f2.y;
            a1[2] = (_Float16)f2.z; a1[3] = (_Float16)f2.w;
            a1[4] = (_Float16)f3.x; a1[5] = (_Float16)f3.y;
            a1[6] = (_Float16)f3.z; a1[7] = (_Float16)f3.w;

            b0[0] = (_Float16)(float)q0.x; b0[1] = (_Float16)(float)q0.y;
            b0[2] = (_Float16)(float)q0.z; b0[3] = (_Float16)(float)q0.w;
            b0[4] = (_Float16)(float)q1.x; b0[5] = (_Float16)(float)q1.y;
            b0[6] = (_Float16)(float)q1.z; b0[7] = (_Float16)(float)q1.w;
            b1[0] = (_Float16)(float)q2.x; b1[1] = (_Float16)(float)q2.y;
            b1[2] = (_Float16)(float)q2.z; b1[3] = (_Float16)(float)q2.w;
            b1[4] = (_Float16)(float)q3.x; b1[5] = (_Float16)(float)q3.y;
            b1[6] = (_Float16)(float)q3.z; b1[7] = (_Float16)(float)q3.w;
        } else {
            // Zero-point fold tile: x column of ones, B column of -zp[n].
            if (lseg == 0) {
                a0[0] = (_Float16)1.0f;
                b0[0] = (_Float16)(-zp[n_base + lrow]);
            }
        }

        __syncthreads();   // previous iteration's fragment reads complete
        *(v8h*)&As[lrow * LDS_PITCH + lseg    ] = a0;
        *(v8h*)&As[lrow * LDS_PITCH + lseg + 8] = a1;
        *(v8h*)&Bs[lrow * LDS_PITCH + lseg    ] = b0;
        *(v8h*)&Bs[lrow * LDS_PITCH + lseg + 8] = b1;
        __syncthreads();   // tile visible to all waves

        // Fragment loads per documented 16-bit WMMA VGPR layouts.
        v16h a_frag[4], b_frag[2];
        #pragma unroll
        for (int fm = 0; fm < 4; ++fm) {
            const _Float16* base = &As[(m_off + fm * 16 + lm) * LDS_PITCH];
            // A 16x32: lane lh half holds K = [8h,8h+8) and [16+8h,16+8h+8)
            v8h lo = *(const v8h*)(base + 8 * lh);
            v8h hi = *(const v8h*)(base + 16 + 8 * lh);
            a_frag[fm] = __builtin_shufflevector(lo, hi,
                0,1,2,3,4,5,6,7,8,9,10,11,12,13,14,15);
        }
        #pragma unroll
        for (int fn = 0; fn < 2; ++fn) {
            const _Float16* base = &Bs[(n_off + fn * 16 + lm) * LDS_PITCH];
            // B 32x16: lane lh half holds K = [8h,8h+8) and [16+8h,16+8h+8)
            v8h lo = *(const v8h*)(base + 8 * lh);
            v8h hi = *(const v8h*)(base + 16 + 8 * lh);
            b_frag[fn] = __builtin_shufflevector(lo, hi,
                0,1,2,3,4,5,6,7,8,9,10,11,12,13,14,15);
        }

        #pragma unroll
        for (int fm = 0; fm < 4; ++fm)
            #pragma unroll
            for (int fn = 0; fn < 2; ++fn)
                acc[fm][fn] = __builtin_amdgcn_wmma_f32_16x16x32_f16(
                    /*neg_a=*/false, a_frag[fm],
                    /*neg_b=*/false, b_frag[fn],
                    /*c_mod=*/(short)0, acc[fm][fn],
                    /*reuse_a=*/false, /*reuse_b=*/false);
    }

    // Epilogue: out[m][n] = scale[n] * acc. C/D layout: n = lane%16, m = r + 8*(lane/16).
    #pragma unroll
    for (int fn = 0; fn < 2; ++fn) {
        const int   ng = n_base + n_off + fn * 16 + lm;
        const float s  = scale[ng];
        #pragma unroll
        for (int fm = 0; fm < 4; ++fm) {
            #pragma unroll
            for (int r = 0; r < 8; ++r) {
                const int mg = m_base + m_off + fm * 16 + r + 8 * lh;
                out[(size_t)mg * N_DIM + ng] = s * acc[fm][fn][r];
            }
        }
    }
}

extern "C" void kernel_launch(void* const* d_in, const int* in_sizes, int n_in,
                              void* d_out, int out_size, void* d_ws, size_t ws_size,
                              hipStream_t stream) {
    (void)n_in; (void)out_size; (void)d_ws; (void)ws_size;
    const float* x     = (const float*)d_in[0];   // [4,2048,4096] f32
    const int*   wq    = (const int*)  d_in[1];   // [4096,4096] int32 (values 0..126)
    const float* scale = (const float*)d_in[2];   // [4096]
    const float* zp    = (const float*)d_in[3];   // [4096]
    float*       out   = (float*)d_out;           // [4,2048,4096] f32

    const int M = in_sizes[0] / K_DIM;            // 8192, multiple of BM
    dim3 grid(N_DIM / BN, M / BM);                // 32 x 64 blocks
    qlinear_wmma_kernel<<<grid, 256, 0, stream>>>(x, wq, scale, zp, out);
}
